// complex_attn_9251359555967
// MI455X (gfx1250) — matrix operand order repaired and checked by reading, not compiled, not run on hardware
//
#include <hip/hip_runtime.h>

#define SEQQ   4096
#define DIMM   1024
#define KPROJ  256
#define NHEAD  16
#define DHEAD  64
#define NBATCH 4

typedef __bf16 bf16;
typedef __attribute__((ext_vector_type(16))) __bf16 v16bf;
typedef __attribute__((ext_vector_type(8)))  float  v8f;

union Frag16 { uint4 u[2]; v16bf v; };

__device__ __forceinline__ v8f wmma_bf16(v16bf a, v16bf b, v8f c) {
    // D(f32 16x16) = A(bf16 16x32) * B(bf16 32x16) + C
    return __builtin_amdgcn_wmma_f32_16x16x32_bf16(
        /*neg_a=*/false, a, /*neg_b=*/false, b,
        /*c_mod=*/(short)0, c, /*reuse_a=*/false, /*reuse_b=*/false);
}

// A operand (16x32 bf16): lane half lh: elems j<8 -> K=lh*8+j ; j>=8 -> K=16+lh*8+(j-8)
__device__ __forceinline__ v16bf load_afrag(const bf16* rowp, int lh) {
    Frag16 f;
    f.u[0] = *(const uint4*)(rowp + lh * 8);
    f.u[1] = *(const uint4*)(rowp + 16 + lh * 8);
    return f.v;
}
// B operand (32x16 bf16): lane half lh: elems j<8 -> K=lh*8+j ; j>=8 -> K=16+lh*8+(j-8)
__device__ __forceinline__ v16bf load_bfrag(const bf16* colp, int lh) {
    Frag16 f;
    f.u[0] = *(const uint4*)(colp + lh * 8);
    f.u[1] = *(const uint4*)(colp + 16 + lh * 8);
    return f.v;
}

// Convert 16 contiguous f32 -> 16 bf16, store as two b128 to LDS
__device__ __forceinline__ void cvt16(const float* __restrict__ p, bf16* dst) {
    const float4* p4 = (const float4*)p;
    Frag16 f;
#pragma unroll
    for (int i = 0; i < 4; ++i) {
        float4 x = p4[i];
        f.v[4*i+0] = (bf16)x.x; f.v[4*i+1] = (bf16)x.y;
        f.v[4*i+2] = (bf16)x.z; f.v[4*i+3] = (bf16)x.w;
    }
    *(uint4*)(dst)     = f.u[0];
    *(uint4*)(dst + 8) = f.u[1];
}

// LDS byte offset of a shared-memory pointer: per aperture mapping the low
// 32 bits of the flat address ARE the LDS offset (ISA 10.2).
__device__ __forceinline__ unsigned lds_off(const void* p) {
    return (unsigned)(size_t)p;
}
// Per-lane async global->LDS copy of 16 bytes (GV mode; INST_OFFSET advances
// both the LDS and global addresses). Tracked by ASYNCcnt.
__device__ __forceinline__ void async_copy_b128(unsigned lds_addr, const void* gptr, int ofs) {
    asm volatile("global_load_async_to_lds_b128 %0, %1, off offset:%2"
                 :: "v"(lds_addr), "v"((unsigned long long)(size_t)gptr), "i"(ofs)
                 : "memory");
}
__device__ __forceinline__ void wait_asynccnt0() {
    asm volatile("s_wait_asynccnt 0x0" ::: "memory");
}

// ---------------------------------------------------------------------------
// colsum: out[0:256) = sum_n proj_k[n,k] ; out[256:512) = sum_n proj_v[n,k]
// ---------------------------------------------------------------------------
__global__ __launch_bounds__(256) void colsum_kernel(
    const float* __restrict__ pk, const float* __restrict__ pv,
    float* __restrict__ out)
{
    const int k = threadIdx.x;
    const float* p = (blockIdx.x == 0) ? pk : pv;
    float s = 0.0f;
    for (int n = 0; n < SEQQ; ++n) s += p[(size_t)n * KPROJ + k];
    out[blockIdx.x * KPROJ + k] = s;
}

// ---------------------------------------------------------------------------
// GEMM NT:  C[M,N] = A[M,Kd] * B[N,Kd]^T + rowscale[row % rs_mod] * bias[col]
// A fp32 or bf16 (template), B/bias fp32, C fp32 or bf16 (template).
// Tile 128x128, BK=32, 256 threads (8 waves), wave tile 32x64 (2x4 WMMA accs).
// M, N multiples of 128; Kd multiple of 32.
// bf16 A tiles are staged with async global->LDS copies (ASYNCcnt domain).
// ---------------------------------------------------------------------------
template<bool A_BF16, bool OUT_BF16>
__global__ __launch_bounds__(256) void gemm_nt_kernel(
    const void* __restrict__ Ain, const float* __restrict__ Bw,
    const float* __restrict__ bias, const float* __restrict__ rowscale, int rs_mod,
    void* __restrict__ Cout, int M, int N, int Kd)
{
    __shared__ alignas(16) bf16 As[128][40];
    __shared__ alignas(16) bf16 Bs[128][40];

    const int tid  = threadIdx.x;
    const int lane = tid & 31, wave = tid >> 5;
    const int lm = lane & 15,  lh = lane >> 4;
    const int wr = wave >> 1,  wc = wave & 1;
    const int m0 = blockIdx.y * 128, n0 = blockIdx.x * 128;
    const int srow = tid >> 1, skb = (tid & 1) * 16;

    const unsigned ldsA = lds_off(&As[srow][skb]);

    v8f acc[2][4] = {};

    for (int kt = 0; kt < Kd; kt += 32) {
        if (A_BF16) {
            // pure bf16 copy: issue async global->LDS (no VGPR round trip)
            const bf16* ap = (const bf16*)Ain + (size_t)(m0 + srow) * Kd + kt + skb;
            async_copy_b128(ldsA, ap, 0);
            async_copy_b128(ldsA, ap, 16);
        } else {
            cvt16((const float*)Ain + (size_t)(m0 + srow) * Kd + kt + skb, &As[srow][skb]);
        }
        cvt16(Bw + (size_t)(n0 + srow) * Kd + kt + skb, &Bs[srow][skb]);
        if (kt + 32 < Kd) {   // global_prefetch_b8 next B tile
            __builtin_prefetch(Bw + (size_t)(n0 + srow) * Kd + kt + 32 + skb, 0, 1);
        }
        if (A_BF16) wait_asynccnt0();
        __syncthreads();

        v16bf af[2], bfg[4];
#pragma unroll
        for (int rt = 0; rt < 2; ++rt) af[rt]  = load_afrag(&As[wr*32 + rt*16 + lm][0], lh);
#pragma unroll
        for (int ct = 0; ct < 4; ++ct) bfg[ct] = load_bfrag(&Bs[wc*64 + ct*16 + lm][0], lh);
#pragma unroll
        for (int rt = 0; rt < 2; ++rt)
#pragma unroll
            for (int ct = 0; ct < 4; ++ct)
                acc[rt][ct] = wmma_bf16(af[rt], bfg[ct], acc[rt][ct]);
        __syncthreads();
    }

#pragma unroll
    for (int rt = 0; rt < 2; ++rt) {
        const int rbase = m0 + wr*32 + rt*16 + lh*8;
#pragma unroll
        for (int ct = 0; ct < 4; ++ct) {
            const int col = n0 + wc*64 + ct*16 + lm;
            const float bv = bias ? bias[col] : 0.0f;
#pragma unroll
            for (int r = 0; r < 8; ++r) {
                const int row = rbase + r;
                const float rs  = rowscale ? rowscale[row % rs_mod] : 1.0f;
                const float val = acc[rt][ct][r] + rs * bv;
                if (OUT_BF16) ((bf16*)Cout)[(size_t)row * N + col] = (bf16)val;
                else          ((float*)Cout)[(size_t)row * N + col] = val;
            }
        }
    }
}

// ---------------------------------------------------------------------------
// GEMM TN:  C[m,n] = sum_s A[s,m] * B[s,n]   (down-projection, per batch z)
// A = proj [S x Mtot] f32 (shared across batches), B = x[z] [S x Ntot] f32,
// C = [z][Mtot x Ntot] bf16.  Tiles 128x128, BK(S)=32.
// ---------------------------------------------------------------------------
__global__ __launch_bounds__(256) void gemm_tn_kernel(
    const float* __restrict__ Ap, const float* __restrict__ Bp,
    bf16* __restrict__ Cout, int S, int Mtot, int Ntot)
{
    __shared__ alignas(16) bf16 As[128][40];
    __shared__ alignas(16) bf16 Bs[128][40];

    const float* Bb = Bp + (size_t)blockIdx.z * S * Ntot;
    bf16* Cb = Cout + (size_t)blockIdx.z * Mtot * Ntot;

    const int tid  = threadIdx.x;
    const int lane = tid & 31, wave = tid >> 5;
    const int lm = lane & 15,  lh = lane >> 4;
    const int wr = wave >> 1,  wc = wave & 1;
    const int m0 = blockIdx.y * 128, n0 = blockIdx.x * 128;
    const int sj = tid >> 3, si0 = (tid & 7) * 16;

    v8f acc[2][4] = {};

    for (int st = 0; st < S; st += 32) {
        const float4* a4 = (const float4*)(Ap + (size_t)(st + sj) * Mtot + m0 + si0);
#pragma unroll
        for (int i = 0; i < 4; ++i) {   // transposed store: As[m][k]
            float4 x = a4[i];
            As[si0 + 4*i + 0][sj] = (bf16)x.x;
            As[si0 + 4*i + 1][sj] = (bf16)x.y;
            As[si0 + 4*i + 2][sj] = (bf16)x.z;
            As[si0 + 4*i + 3][sj] = (bf16)x.w;
        }
        const float4* b4 = (const float4*)(Bb + (size_t)(st + sj) * Ntot + n0 + si0);
#pragma unroll
        for (int i = 0; i < 4; ++i) {   // transposed store: Bs[n][k]
            float4 x = b4[i];
            Bs[si0 + 4*i + 0][sj] = (bf16)x.x;
            Bs[si0 + 4*i + 1][sj] = (bf16)x.y;
            Bs[si0 + 4*i + 2][sj] = (bf16)x.z;
            Bs[si0 + 4*i + 3][sj] = (bf16)x.w;
        }
        __syncthreads();

        v16bf af[2], bfg[4];
#pragma unroll
        for (int rt = 0; rt < 2; ++rt) af[rt]  = load_afrag(&As[wr*32 + rt*16 + lm][0], lh);
#pragma unroll
        for (int ct = 0; ct < 4; ++ct) bfg[ct] = load_bfrag(&Bs[wc*64 + ct*16 + lm][0], lh);
#pragma unroll
        for (int rt = 0; rt < 2; ++rt)
#pragma unroll
            for (int ct = 0; ct < 4; ++ct)
                acc[rt][ct] = wmma_bf16(af[rt], bfg[ct], acc[rt][ct]);
        __syncthreads();
    }

#pragma unroll
    for (int rt = 0; rt < 2; ++rt) {
        const int rbase = m0 + wr*32 + rt*16 + lh*8;
#pragma unroll
        for (int ct = 0; ct < 4; ++ct) {
            const int col = n0 + wc*64 + ct*16 + lm;
#pragma unroll
            for (int r = 0; r < 8; ++r)
                Cb[(size_t)(rbase + r) * Ntot + col] = (bf16)acc[rt][ct][r];
        }
    }
}

// ---------------------------------------------------------------------------
// Fused Linformer attention (no softmax):
//   out[b,n,hd*64+d] = sum_k ( (Q[b,n,:64]·Kp[b,k,:64]) * 0.125 ) * Vp[b,k,d]
// Grid (SEQ/128, NHEAD, NBATCH), 256 threads; Kp & Vp^T staged in LDS;
// each wave does 16 Q rows: dots 16x32 at a time -> LDS relayout -> out accum.
// Kp tile staged via async global->LDS copies.
// ---------------------------------------------------------------------------
__global__ __launch_bounds__(256) void attn_kernel(
    const bf16* __restrict__ q, const bf16* __restrict__ kp,
    const bf16* __restrict__ vp, bf16* __restrict__ attn)
{
    __shared__ alignas(16) bf16 Ks[KPROJ][72];     // [k][d]
    __shared__ alignas(16) bf16 Vs[DHEAD][264];    // [d][k] (transposed)
    __shared__ alignas(16) bf16 Ds[8][16][40];     // per-wave dots scratch

    const int tid  = threadIdx.x;
    const int lane = tid & 31, wave = tid >> 5;
    const int lm = lane & 15,  lh = lane >> 4;
    const int b = blockIdx.z, hd = blockIdx.y;
    const int m0 = blockIdx.x * 128 + wave * 16;

    const bf16* kg = kp + (size_t)b * KPROJ * DIMM + hd * DHEAD;
    const bf16* vg = vp + (size_t)b * KPROJ * DIMM + hd * DHEAD;
    {
        // Kp: straight bf16 copy, one k-row per thread -> async to LDS
        const bf16* kr = kg + (size_t)tid * DIMM;
        const unsigned ldsK = lds_off(&Ks[tid][0]);
#pragma unroll
        for (int c = 0; c < DHEAD; c += 8)
            async_copy_b128(ldsK, kr, c * (int)sizeof(bf16));
        // Vp: transposed stage (needs element shuffle -> VGPR path)
        const bf16* vr = vg + (size_t)tid * DIMM;
#pragma unroll
        for (int c = 0; c < DHEAD; ++c)
            Vs[c][tid] = vr[c];
    }
    wait_asynccnt0();
    __syncthreads();

    const bf16* qrow = q + ((size_t)b * SEQQ + m0 + lm) * DIMM + hd * DHEAD;
    v16bf qf[2];
#pragma unroll
    for (int dt = 0; dt < 2; ++dt) qf[dt] = load_afrag(qrow + dt * 32, lh);

    v8f oacc[4] = {};
#pragma unroll
    for (int kt = 0; kt < 8; ++kt) {               // k tiles of 32 (K=256)
        v8f dacc[2] = {};
#pragma unroll
        for (int kh = 0; kh < 2; ++kh)
#pragma unroll
            for (int dt = 0; dt < 2; ++dt) {
                v16bf kf = load_bfrag(&Ks[kt*32 + kh*16 + lm][dt*32], lh);
                dacc[kh] = wmma_bf16(qf[dt], kf, dacc[kh]);
            }
        // scale, spill C-fragment to per-wave LDS scratch as bf16
#pragma unroll
        for (int kh = 0; kh < 2; ++kh)
#pragma unroll
            for (int r = 0; r < 8; ++r)
                Ds[wave][lh*8 + r][kh*16 + lm] = (bf16)(dacc[kh][r] * 0.125f);
        // reload as A-fragment (16x32)
        Frag16 af;
        af.u[0] = *(const uint4*)&Ds[wave][lm][lh*8];
        af.u[1] = *(const uint4*)&Ds[wave][lm][16 + lh*8];
#pragma unroll
        for (int ct = 0; ct < 4; ++ct) {
            v16bf vf = load_bfrag(&Vs[ct*16 + lm][kt*32], lh);
            oacc[ct] = wmma_bf16(af.v, vf, oacc[ct]);
        }
    }

    bf16* op = attn + (size_t)b * SEQQ * DIMM + hd * DHEAD;
#pragma unroll
    for (int ct = 0; ct < 4; ++ct)
#pragma unroll
        for (int r = 0; r < 8; ++r)
            op[(size_t)(m0 + lh*8 + r) * DIMM + ct*16 + lm] = (bf16)oacc[ct][r];
}

// ---------------------------------------------------------------------------
extern "C" void kernel_launch(void* const* d_in, const int* in_sizes, int n_in,
                              void* d_out, int out_size, void* d_ws, size_t ws_size,
                              hipStream_t stream)
{
    (void)in_sizes; (void)n_in; (void)out_size; (void)ws_size;
    const float* xq = (const float*)d_in[0];
    const float* xk = (const float*)d_in[1];
    const float* xv = (const float*)d_in[2];
    const float* Wq = (const float*)d_in[3];
    const float* bq = (const float*)d_in[4];
    const float* Wk = (const float*)d_in[5];
    const float* bk = (const float*)d_in[6];
    const float* Wv = (const float*)d_in[7];
    const float* bv = (const float*)d_in[8];
    const float* pk = (const float*)d_in[9];
    const float* pv = (const float*)d_in[10];
    const float* Wo = (const float*)d_in[11];
    const float* bo = (const float*)d_in[12];
    float* out = (float*)d_out;

    const size_t BSD = (size_t)NBATCH * SEQQ * DIMM;   // 16M elems
    const size_t BKD = (size_t)NBATCH * KPROJ * DIMM;  // 1M elems
    bf16* qb    = (bf16*)d_ws;        // Q, bf16            [B*SEQ, DIM]
    bf16* xkp   = qb    + BSD;        // proj_k^T @ xk      [B*256, DIM]
    bf16* xvp   = xkp   + BKD;        // proj_v^T @ xv      [B*256, DIM]
    bf16* keysp = xvp   + BKD;        // down-projected K   [B*256, DIM]
    bf16* valsp = keysp + BKD;        // down-projected V   [B*256, DIM]
    bf16* attnb = valsp + BKD;        // attention output   [B*SEQ, DIM]
    float* sums = (float*)(attnb + BSD);  // 512 f32: colsum(proj_k), colsum(proj_v)

    // proj column sums (for commuted bias term)
    colsum_kernel<<<dim3(2), dim3(256), 0, stream>>>(pk, pv, sums);
    // down-project sequence dimension FIRST (commutes with the linear layers)
    gemm_tn_kernel<<<dim3(DIMM/128, KPROJ/128, NBATCH), 256, 0, stream>>>(pk, xk, xkp, SEQQ, KPROJ, DIMM);
    gemm_tn_kernel<<<dim3(DIMM/128, KPROJ/128, NBATCH), 256, 0, stream>>>(pv, xv, xvp, SEQQ, KPROJ, DIMM);
    // Q projection on full sequence
    gemm_nt_kernel<false, true><<<dim3(DIMM/128, (NBATCH*SEQQ)/128), 256, 0, stream>>>(
        xq, Wq, bq, nullptr, 1, qb, NBATCH*SEQQ, DIMM, DIMM);
    // K/V projections on down-projected (256-row) activations; bias scaled by colsum(proj)
    gemm_nt_kernel<true, true><<<dim3(DIMM/128, (NBATCH*KPROJ)/128), 256, 0, stream>>>(
        xkp, Wk, bk, sums, KPROJ, keysp, NBATCH*KPROJ, DIMM, DIMM);
    gemm_nt_kernel<true, true><<<dim3(DIMM/128, (NBATCH*KPROJ)/128), 256, 0, stream>>>(
        xvp, Wv, bv, sums + KPROJ, KPROJ, valsp, NBATCH*KPROJ, DIMM, DIMM);
    // fused attention
    attn_kernel<<<dim3(SEQQ/128, NHEAD, NBATCH), 256, 0, stream>>>(qb, keysp, valsp, attnb);
    // output projection
    gemm_nt_kernel<true, false><<<dim3(DIMM/128, (NBATCH*SEQQ)/128), 256, 0, stream>>>(
        attnb, Wo, bo, nullptr, 1, out, NBATCH*SEQQ, DIMM, DIMM);
}
